// FlashCausalGroupedQueryPageAttention_55482387530053
// MI455X (gfx1250) — matrix operand order repaired and checked by reading, not compiled, not run on hardware
//
#include <hip/hip_runtime.h>
#include <cmath>

// Flash causal GQA attention for MI455X (gfx1250, wave32, WMMA bf16).
// The paged-cache scatter+gather in the reference is an identity, so we read
// K/V contiguously and never touch the caches (inputs must not be mutated).
//
// v2: software-pipelined global->LDS staging (next chunk loaded into registers
//     during compute), batched LDS fragment loads ahead of WMMA groups,
//     softmax scale folded into the Q fragments.

#define B_    4
#define S_    1024
#define NQO_  32
#define NKV_  8
#define D_    128
#define G_    4
#define SCALE_ 0.08838834764831845f  // 1/sqrt(128)

typedef __attribute__((ext_vector_type(16))) __bf16 v16bf;
typedef __attribute__((ext_vector_type(8)))  __bf16 v8bf;
typedef __attribute__((ext_vector_type(4)))  __bf16 v4bf;
typedef __attribute__((ext_vector_type(8)))  float  v8f;

// LDS row strides (in bf16 elements), padded to avoid bank conflicts.
#define KSTR 136   // K tile: 32 rows (kv) x 128 (d)
#define VSTR 40    // V^T tile: 128 rows (d) x 32 (kv)
#define PSTR 40    // P tile per wave: 16 rows (q) x 32 (kv)

__device__ __forceinline__ v16bf ld_bfrag16(const __bf16* p) {
  // 32 contiguous bytes as two 16B LDS loads, combined into one fragment.
  v8bf x0 = *(const v8bf*)p;
  v8bf x1 = *(const v8bf*)(p + 8);
  v16bf r;
#pragma unroll
  for (int i = 0; i < 8; ++i) { r[i] = x0[i]; r[i + 8] = x1[i]; }
  return r;
}

__device__ __forceinline__ v4bf cvt4(float a, float b, float c, float d) {
  v4bf r;
  r[0] = (__bf16)a; r[1] = (__bf16)b; r[2] = (__bf16)c; r[3] = (__bf16)d;
  return r;
}

__global__ __launch_bounds__(128)
void fa_gqa_prefill_kernel(const float* __restrict__ q,
                           const float* __restrict__ k,
                           const float* __restrict__ v,
                           float* __restrict__ out)
{
  __shared__ __bf16 k_lds[32 * KSTR];     // K chunk, row-major [kv][d]
  __shared__ __bf16 v_lds[128 * VSTR];    // V chunk, transposed [d][kv]
  __shared__ __bf16 p_lds[4 * 16 * PSTR]; // per-wave P transpose staging

  const int tid  = threadIdx.x;
  const int lane = tid & 31;
  const int wav  = tid >> 5;             // 0..3 == g (query head within group)
  const int nqt  = S_ / 16;              // 64 q-tiles per sequence

  const int wg = blockIdx.x;
  const int qt = wg % nqt;
  const int h  = (wg / nqt) % NKV_;
  const int b  = wg / (nqt * NKV_);
  const int q0 = qt * 16;
  const int qh = h * G_ + wav;           // global query head
  const size_t bS = (size_t)b * S_;

  const int m     = lane & 15;           // column / row-in-tile index
  const int hlf   = lane >> 4;           // 0: lanes 0-15, 1: lanes 16-31
  const int rbase = hlf * 8;             // C-layout row base for this lane

  // ---------------- Q A-fragments (scale folded in): 4 chunks of K=32 ------
  // A fragment as loaded here (contiguous k, not AMD's table order): lane<16 holds row m, d = {c+0..15};
  // lane>=16 holds d = {c+16..31}.
  v16bf aq[4];
  {
    const float* qrow = q + (bS + q0 + m) * (NQO_ * D_) + qh * D_;
    const int lo = hlf * 16;
#pragma unroll
    for (int c = 0; c < 4; ++c) {
      float4 f0 = *(const float4*)(qrow + 32 * c + lo);
      float4 f1 = *(const float4*)(qrow + 32 * c + lo + 4);
      float4 f2 = *(const float4*)(qrow + 32 * c + lo + 8);
      float4 f3 = *(const float4*)(qrow + 32 * c + lo + 12);
      v16bf a;
      a[0] = (__bf16)(f0.x*SCALE_); a[1] = (__bf16)(f0.y*SCALE_);
      a[2] = (__bf16)(f0.z*SCALE_); a[3] = (__bf16)(f0.w*SCALE_);
      a[4] = (__bf16)(f1.x*SCALE_); a[5] = (__bf16)(f1.y*SCALE_);
      a[6] = (__bf16)(f1.z*SCALE_); a[7] = (__bf16)(f1.w*SCALE_);
      a[8] = (__bf16)(f2.x*SCALE_); a[9] = (__bf16)(f2.y*SCALE_);
      a[10]= (__bf16)(f2.z*SCALE_); a[11]= (__bf16)(f2.w*SCALE_);
      a[12]= (__bf16)(f3.x*SCALE_); a[13]= (__bf16)(f3.y*SCALE_);
      a[14]= (__bf16)(f3.z*SCALE_); a[15]= (__bf16)(f3.w*SCALE_);
      aq[c] = a;
    }
  }

  // ---------------- flash-attention state ----------------------------------
  v8f o[8];                              // O accumulator: 8 d-tiles of 16x16
#pragma unroll
  for (int i = 0; i < 8; ++i) o[i] = (v8f){0.f,0.f,0.f,0.f,0.f,0.f,0.f,0.f};
  float mrow[8], lrow[8];
#pragma unroll
  for (int r = 0; r < 8; ++r) { mrow[r] = -INFINITY; lrow[r] = 0.f; }

  __bf16* pw = p_lds + wav * 16 * PSTR;

  // ---------------- pipelined global->register staging ---------------------
  const int tK  = tid >> 2;              // K: token 0..31
  const int d0K = (tid & 3) * 32;        // K: 32-wide d segment
  const int t0V = (tid >> 4) * 4;        // V: 4 tokens
  const int d0V = (tid & 15) * 8;        // V: 8 d values

  float kstg[32];                        // K[tK][d0K..d0K+31]
  float vstg[4][8];                      // V[t0V+j][d0V..d0V+7]

  auto load_chunk = [&](int kv0) {
    const float* krow = k + (bS + kv0 + tK) * (NKV_ * D_) + h * D_ + d0K;
#pragma unroll
    for (int j = 0; j < 8; ++j)
      *(float4*)&kstg[4 * j] = *(const float4*)(krow + 4 * j);
#pragma unroll
    for (int j = 0; j < 4; ++j) {
      const float* vrow = v + (bS + kv0 + t0V + j) * (NKV_ * D_) + h * D_ + d0V;
      *(float4*)&vstg[j][0] = *(const float4*)(vrow);
      *(float4*)&vstg[j][4] = *(const float4*)(vrow + 4);
    }
  };

  const int kv_need = q0 + 16;           // causal: columns 0 .. q0+15
  load_chunk(0);

  for (int kv0 = 0; kv0 < kv_need; kv0 += 32) {
    __syncthreads();                     // previous iteration's LDS reads done

    // ---- drain staging registers into LDS (fp32 -> bf16) ------------------
#pragma unroll
    for (int j = 0; j < 8; ++j)
      *(v4bf*)(k_lds + tK * KSTR + d0K + 4 * j) =
          cvt4(kstg[4*j], kstg[4*j+1], kstg[4*j+2], kstg[4*j+3]);
#pragma unroll
    for (int i = 0; i < 8; ++i)
      *(v4bf*)(v_lds + (d0V + i) * VSTR + t0V) =
          cvt4(vstg[0][i], vstg[1][i], vstg[2][i], vstg[3][i]);

    __syncthreads();                     // LDS tiles ready

    // ---- issue next chunk's global loads; latency hides under compute -----
    if (kv0 + 32 < kv_need) load_chunk(kv0 + 32);

    // ---- scores: S[16q x 32k] = Q * K^T (8 WMMAs, batched frag loads) -----
    // B layout (32d x 16k): lane holds column k=m, d = hlf*16 + {0..15} + 32c.
    v8f s0 = (v8f){0.f,0.f,0.f,0.f,0.f,0.f,0.f,0.f};
    v8f s1 = (v8f){0.f,0.f,0.f,0.f,0.f,0.f,0.f,0.f};
    {
      const int dbase = hlf * 16;
      v16bf kf[4];
#pragma unroll
      for (int c = 0; c < 4; ++c)
        kf[c] = ld_bfrag16(k_lds + m * KSTR + 32 * c + dbase);
#pragma unroll
      for (int c = 0; c < 4; ++c)
        s0 = __builtin_amdgcn_wmma_f32_16x16x32_bf16(false, aq[c], false, kf[c],
                                                     (short)0, s0, false, false);
#pragma unroll
      for (int c = 0; c < 4; ++c)
        kf[c] = ld_bfrag16(k_lds + (16 + m) * KSTR + 32 * c + dbase);
#pragma unroll
      for (int c = 0; c < 4; ++c)
        s1 = __builtin_amdgcn_wmma_f32_16x16x32_bf16(false, aq[c], false, kf[c],
                                                     (short)0, s1, false, false);
    }

    // ---- causal mask + online softmax (C layout; scale already folded) ----
    float corr[8];
#pragma unroll
    for (int r = 0; r < 8; ++r) {
      const int qpos = q0 + rbase + r;
      float x0 = (kv0 + m      <= qpos) ? s0[r] : -INFINITY;
      float x1 = (kv0 + 16 + m <= qpos) ? s1[r] : -INFINITY;
      float mx = fmaxf(x0, x1);
#pragma unroll
      for (int off = 1; off <= 8; off <<= 1)
        mx = fmaxf(mx, __shfl_xor(mx, off, 32));   // 16-lane row reduction
      const float mnew = fmaxf(mrow[r], mx);
      const float cr   = __expf(mrow[r] - mnew);   // first iter: exp(-inf)=0
      float p0 = __expf(x0 - mnew);
      float p1 = __expf(x1 - mnew);
      float ps = p0 + p1;
#pragma unroll
      for (int off = 1; off <= 8; off <<= 1)
        ps += __shfl_xor(ps, off, 32);
      lrow[r] = lrow[r] * cr + ps;
      mrow[r] = mnew;
      corr[r] = cr;
      // stage P (bf16) for C-layout -> A-layout transpose
      pw[(rbase + r) * PSTR + m]      = (__bf16)p0;
      pw[(rbase + r) * PSTR + 16 + m] = (__bf16)p1;
    }

    // rescale O accumulator
#pragma unroll
    for (int nt = 0; nt < 8; ++nt)
#pragma unroll
      for (int r = 0; r < 8; ++r) o[nt][r] *= corr[r];

    // same-wave LDS ordering for the P transpose round-trip
    asm volatile("s_wait_dscnt 0" ::: "memory");

    // P as A-fragment (16q x 32kv)
    v16bf pf;
    {
      const int c0 = hlf * 16;
      v8bf x0 = *(const v8bf*)(pw + m * PSTR + c0);
      v8bf x1 = *(const v8bf*)(pw + m * PSTR + c0 + 8);
#pragma unroll
      for (int i = 0; i < 8; ++i) { pf[i] = x0[i]; pf[i + 8] = x1[i]; }
    }

    // ---- O += P * V (8 WMMAs over d-tiles, batched frag loads) ------------
#pragma unroll
    for (int g4 = 0; g4 < 2; ++g4) {
      v16bf vf[4];
#pragma unroll
      for (int i = 0; i < 4; ++i)
        vf[i] = ld_bfrag16(v_lds + ((g4 * 4 + i) * 16 + m) * VSTR + hlf * 16);
#pragma unroll
      for (int i = 0; i < 4; ++i)
        o[g4 * 4 + i] = __builtin_amdgcn_wmma_f32_16x16x32_bf16(
            false, pf, false, vf[i], (short)0, o[g4 * 4 + i], false, false);
    }
  }

  // ---------------- epilogue: normalize and store --------------------------
  float inv[8];
#pragma unroll
  for (int r = 0; r < 8; ++r) inv[r] = 1.0f / lrow[r];
#pragma unroll
  for (int r = 0; r < 8; ++r) {
    float* orow = out + (bS + q0 + rbase + r) * (NQO_ * D_) + qh * D_ + m;
#pragma unroll
    for (int nt = 0; nt < 8; ++nt) orow[nt * 16] = o[nt][r] * inv[r];
  }
}

extern "C" void kernel_launch(void* const* d_in, const int* in_sizes, int n_in,
                              void* d_out, int out_size, void* d_ws, size_t ws_size,
                              hipStream_t stream) {
  const float* q = (const float*)d_in[0];  // query  [B*S, NQO*D]
  const float* k = (const float*)d_in[1];  // key    [B*S, NKV*D]
  const float* v = (const float*)d_in[2];  // value  [B*S, NKV*D]
  // d_in[3..6]: key_cache / value_cache / new_cache_slots / block_tables —
  // the scatter+gather through the paged cache is an identity for the output,
  // and inputs must not be mutated, so they are intentionally unused.
  float* out = (float*)d_out;

  dim3 grid(B_ * NKV_ * (S_ / 16));  // 2048 workgroups
  dim3 block(128);                   // 4 waves: the 4 GQA heads per KV head
  hipLaunchKernelGGL(fa_gqa_prefill_kernel, grid, block, 0, stream, q, k, v, out);

  (void)in_sizes; (void)n_in; (void)out_size; (void)d_ws; (void)ws_size;
}
